// MultiHeadedMedicalAxialSelfAttentionWrapper_13082470383835
// MI455X (gfx1250) — matrix operand order repaired and checked by reading, not compiled, not run on hardware
//
#include <hip/hip_runtime.h>

typedef _Float16 v16h __attribute__((ext_vector_type(16)));
typedef _Float16 h8   __attribute__((ext_vector_type(8)));
typedef float    v8f  __attribute__((ext_vector_type(8)));
typedef float    v2f  __attribute__((ext_vector_type(2)));

#define CLN   96
#define HSZ   64
#define WSZ   64
#define DIMV  64
#define FEAT  (CLN*HSZ*WSZ)   /* 393216 */
#define EPSV  1e-5f
#define QS    100             /* s_qkv row stride (floats), 16B aligned, padded */
#define SS    104             /* sim strip row stride (halfs), 16B aligned      */

__device__ __forceinline__ v16h pack16(float4 a, float4 b, float4 c, float4 d) {
  v16h r;
  r[0]=(_Float16)a.x;  r[1]=(_Float16)a.y;  r[2]=(_Float16)a.z;  r[3]=(_Float16)a.w;
  r[4]=(_Float16)b.x;  r[5]=(_Float16)b.y;  r[6]=(_Float16)b.z;  r[7]=(_Float16)b.w;
  r[8]=(_Float16)c.x;  r[9]=(_Float16)c.y;  r[10]=(_Float16)c.z; r[11]=(_Float16)c.w;
  r[12]=(_Float16)d.x; r[13]=(_Float16)d.y; r[14]=(_Float16)d.z; r[15]=(_Float16)d.w;
  return r;
}

__global__ __launch_bounds__(128)
void axial_attn(const float* __restrict__ x,
                const float* __restrict__ w_qkv,
                const float* __restrict__ qkv_gamma, const float* __restrict__ qkv_beta,
                const float* __restrict__ qkv_mean,  const float* __restrict__ qkv_var,
                const float* __restrict__ sim_gamma, const float* __restrict__ sim_var,
                const float* __restrict__ out_gamma, const float* __restrict__ out_beta,
                const float* __restrict__ out_mean,  const float* __restrict__ out_var,
                float* __restrict__ out)
{
  __shared__ alignas(16) float    s_qkv[128*QS];        /* 51.2 KB */
  __shared__ alignas(16) _Float16 s_sim[4][16*SS];      /* 13.3 KB, per-wave strips */

  const int b    = blockIdx.x;            /* b = nt*64 + hs, 2048 blocks */
  const int nt   = b >> 6;
  const int hs   = b & 63;
  const int tid  = (int)threadIdx.x;
  const int wave = tid >> 5;
  const int lane = tid & 31;
  const int m16  = lane & 15;
  const int hi   = lane >> 4;

  const float* xb = x + (size_t)nt*FEAT + hs*WSZ;   /* xf[b][ws][cl] = xb[cl*4096 + ws] */

  /* ================= Phase 1: qkv(128x96) = BN( W(128x64) @ X(64x96) ) ============ */
  /* A-frags (W, f32->f16): per ISA 16-bit A layout, halfs 0..7 <-> K=k0+8hi+0..7,
     halfs 8..15 <-> K=k0+16+8hi+0..7 : two pairs of float4 loads per fragment.      */
  v16h afrag[2][2];
#pragma unroll
  for (int t = 0; t < 2; ++t) {
    const int m0 = wave*32 + t*16;
    const float* wr = w_qkv + (m0 + m16)*DIMV;
#pragma unroll
    for (int ks = 0; ks < 2; ++ks) {
      const float* p = wr + ks*32 + 8*hi;
      float4 w0 = *(const float4*)(p + 0);
      float4 w1 = *(const float4*)(p + 4);
      float4 w2 = *(const float4*)(p + 16);
      float4 w3 = *(const float4*)(p + 20);
      afrag[t][ks] = pack16(w0, w1, w2, w3);
    }
  }

  float bnsc[2][8], bnbi[2][8];
#pragma unroll
  for (int t = 0; t < 2; ++t)
#pragma unroll
    for (int r = 0; r < 8; ++r) {
      const int o = wave*32 + t*16 + r + 8*hi;
      const float sc = qkv_gamma[o] * rsqrtf(qkv_var[o] + EPSV);
      bnsc[t][r] = sc;
      bnbi[t][r] = qkv_beta[o] - qkv_mean[o]*sc;
    }

  for (int n = 0; n < 6; ++n) {
    const int n0 = n*16;
    /* B-frags (X, f32->f16): lane holds N=cl=n0+m16, halfs 0..7 <-> ws=8hi+0..7, 8..15 <-> ws=16+8hi+0..7. */
    v16h bfrag[2];
#pragma unroll
    for (int ks = 0; ks < 2; ++ks) {
      const float* p = xb + (n0 + m16)*(HSZ*WSZ) + ks*32 + 8*hi;
      float4 x0 = *(const float4*)(p + 0);
      float4 x1 = *(const float4*)(p + 4);
      float4 x2 = *(const float4*)(p + 16);
      float4 x3 = *(const float4*)(p + 20);
      bfrag[ks] = pack16(x0, x1, x2, x3);
    }
#pragma unroll
    for (int t = 0; t < 2; ++t) {
      v8f acc = {0.f,0.f,0.f,0.f,0.f,0.f,0.f,0.f};
      acc = __builtin_amdgcn_wmma_f32_16x16x32_f16(false, afrag[t][0], false, bfrag[0],
                                                   (short)0, acc, false, false);
      acc = __builtin_amdgcn_wmma_f32_16x16x32_f16(false, afrag[t][1], false, bfrag[1],
                                                   (short)0, acc, false, false);
      const int m0 = wave*32 + t*16;
#pragma unroll
      for (int r = 0; r < 8; ++r)
        s_qkv[(m0 + r + 8*hi)*QS + n0 + m16] = acc[r]*bnsc[t][r] + bnbi[t][r];
    }
  }
  __syncthreads();

  /* ============ Phase 2+3 per group: qk (f32 WMMA K=4) -> softmax -> sv (f16 WMMA) == */
  _Float16* strip = &s_sim[wave][0];
  const int c = m16;                       /* V/output channel-within-group, valid < 8 */

  for (int gg = 0; gg < 2; ++gg) {
    const int g = wave*2 + gg;
    const float* qm = s_qkv + (16*g + 0)*QS;
    const float* km = s_qkv + (16*g + 4)*QS;
    const float* vm = s_qkv + (16*g + 8)*QS;

    /* K-matrix B-frags for qk: f32 4x16, VGPR r <-> c = 2*hi + r */
    v2f kb[6];
#pragma unroll
    for (int jt = 0; jt < 6; ++jt) {
      kb[jt].x = km[(2*hi + 0)*QS + jt*16 + m16];
      kb[jt].y = km[(2*hi + 1)*QS + jt*16 + m16];
    }

    /* V B-frags for sv: B[j][c] = v[c][j], lane N=c, halfs 0..7 <-> j=8hi+0..7, 8..15 <-> j=16+8hi+0..7 (f16) */
    v16h vb[3];
#pragma unroll
    for (int ks = 0; ks < 3; ++ks) {
      v16h vv;
#pragma unroll
      for (int q = 0; q < 16; ++q) vv[q] = (_Float16)0.f;
      if (c < 8) {
        const float* p = vm + c*QS + ks*32 + 8*hi;
        float4 v0 = *(const float4*)(p + 0);
        float4 v1 = *(const float4*)(p + 4);
        float4 v2 = *(const float4*)(p + 16);
        float4 v3 = *(const float4*)(p + 20);
        vv = pack16(v0, v1, v2, v3);
      }
      vb[ks] = vv;
    }

    /* sim BN: additive part is constant per group => cancels in softmax; keep scale */
    const float simsc = sim_gamma[g] * rsqrtf(sim_var[g] + EPSV);
    const int d = 8*g + (c & 7);
    const float osc = out_gamma[d] * rsqrtf(out_var[d] + EPSV);
    const float obi = out_beta[d] - out_mean[d]*osc;
    float* ob = out + (size_t)nt*FEAT + d*HSZ + hs;

    for (int it = 0; it < 6; ++it) {
      const int i0 = it*16;
      /* Q A-frag: A[i][c] = q[c][i]; lane M=i, VGPR r <-> c = 2*hi + r */
      v2f qa;
      qa.x = qm[(2*hi + 0)*QS + i0 + m16];
      qa.y = qm[(2*hi + 1)*QS + i0 + m16];

      v8f acc[6];
#pragma unroll
      for (int jt = 0; jt < 6; ++jt) {
        v8f z = {0.f,0.f,0.f,0.f,0.f,0.f,0.f,0.f};
#if __has_builtin(__builtin_amdgcn_wmma_f32_16x16x4_f32)
        acc[jt] = __builtin_amdgcn_wmma_f32_16x16x4_f32(false, qa, false, kb[jt],
                                                        (short)0, z, false, false);
#else
        /* exact scalar fallback in C/D register layout */
#pragma unroll
        for (int r = 0; r < 8; ++r) {
          const int i = i0 + r + 8*hi;
          const int j = jt*16 + m16;
          float s = 0.f;
#pragma unroll
          for (int cc = 0; cc < 4; ++cc) s += qm[cc*QS + i]*km[cc*QS + j];
          z[r] = s;
        }
        acc[jt] = z;
#endif
      }

#pragma unroll
      for (int jt = 0; jt < 6; ++jt)
#pragma unroll
        for (int r = 0; r < 8; ++r) acc[jt][r] *= simsc;

      /* softmax over j: row i = i0 + r + 8*hi lives in one 16-lane half-wave */
#pragma unroll
      for (int r = 0; r < 8; ++r) {
        float mx = acc[0][r];
#pragma unroll
        for (int jt = 1; jt < 6; ++jt) mx = fmaxf(mx, acc[jt][r]);
#pragma unroll
        for (int msk = 1; msk <= 8; msk <<= 1) mx = fmaxf(mx, __shfl_xor(mx, msk, 32));
        float sum = 0.f;
#pragma unroll
        for (int jt = 0; jt < 6; ++jt) { float e = __expf(acc[jt][r] - mx); acc[jt][r] = e; sum += e; }
#pragma unroll
        for (int msk = 1; msk <= 8; msk <<= 1) sum += __shfl_xor(sum, msk, 32);
        const float inv = 1.0f / sum;
#pragma unroll
        for (int jt = 0; jt < 6; ++jt)
          strip[(r + 8*hi)*SS + jt*16 + m16] = (_Float16)(acc[jt][r]*inv);
      }
      /* same-wave LDS ops are in-order: strip writes visible to reads below */

      /* sv tile: D[i_local][c] = sum_j sim[i][j] * v[c][j] */
      v8f acc2 = {0.f,0.f,0.f,0.f,0.f,0.f,0.f,0.f};
#pragma unroll
      for (int ks = 0; ks < 3; ++ks) {
        const _Float16* sp = strip + m16*SS + ks*32 + 8*hi;
        h8 lo = *(const h8*)(sp);
        h8 hh = *(const h8*)(sp + 16);
        v16h sa = __builtin_shufflevector(lo, hh, 0,1,2,3,4,5,6,7,8,9,10,11,12,13,14,15);
        acc2 = __builtin_amdgcn_wmma_f32_16x16x32_f16(false, sa, false, vb[ks],
                                                      (short)0, acc2, false, false);
      }

      if (c < 8) {
#pragma unroll
        for (int r = 0; r < 8; ++r) {
          const int i = i0 + r + 8*hi;                 /* cl index */
          ob[(size_t)i*(DIMV*HSZ)] = acc2[r]*osc + obi;
        }
      }
    }
  }
}

extern "C" void kernel_launch(void* const* d_in, const int* in_sizes, int n_in,
                              void* d_out, int out_size, void* d_ws, size_t ws_size,
                              hipStream_t stream) {
  (void)in_sizes; (void)n_in; (void)out_size; (void)d_ws; (void)ws_size;
  const float* x         = (const float*)d_in[0];
  const float* w_qkv     = (const float*)d_in[1];
  const float* qkv_gamma = (const float*)d_in[2];
  const float* qkv_beta  = (const float*)d_in[3];
  const float* qkv_mean  = (const float*)d_in[4];
  const float* qkv_var   = (const float*)d_in[5];
  const float* sim_gamma = (const float*)d_in[6];
  /* d_in[7] sim_beta, d_in[8] sim_mean: cancel exactly in softmax */
  const float* sim_var   = (const float*)d_in[9];
  const float* out_gamma = (const float*)d_in[10];
  const float* out_beta  = (const float*)d_in[11];
  const float* out_mean  = (const float*)d_in[12];
  const float* out_var   = (const float*)d_in[13];
  float* out = (float*)d_out;

  dim3 grid(32*64);   /* one block per (nt, hs) */
  dim3 block(128);    /* 4 wave32 waves: 8 QKV M-tiles, 8 attention groups */
  hipLaunchKernelGGL(axial_attn, grid, block, 0, stream,
                     x, w_qkv, qkv_gamma, qkv_beta, qkv_mean, qkv_var,
                     sim_gamma, sim_var, out_gamma, out_beta, out_mean, out_var, out);
}
